// ClsHeadModelMQDD_19799799235005
// MI455X (gfx1250) — compile-verified, matrix operand order checked by reading, not run on hardware
//
#include <hip/hip_runtime.h>
#include <math.h>

// ============================================================================
// MQDD ClsHead model on gfx1250 (MI455X), bf16 WMMA pipeline.
//
// Combined batch: both encodes share weights -> process 4 sequences of 2048
// tokens as M = 8192 rows. Activations are f32 in the workspace; conversion
// to bf16 happens when staging tiles into LDS for WMMA.
//
// All WMMA operands are stored K-contiguous per lane in LDS (A row-major,
// B transposed N-major) so every fragment load is two ds_load_b128.
//
// d_in ordering assumption (jax pytree flatten = alphabetical dict keys):
//  0 input_1(int,2x1x2048) 1 input_2
//  2 emb_ln_b 3 emb_ln_s 4 head_b 5 head_w
//  6 b1 7 b2 8 bk 9 bo 10 bq 11 bv
// 12 ln1_b 13 ln1_s 14 ln2_b 15 ln2_s
// 16 w1 17 w2 18 wk 19 wo 20 wq 21 wv
// 22 out_w 23 pool_b 24 pool_w 25 pos_emb 26 type_emb 27 word_emb
// ============================================================================

typedef __attribute__((ext_vector_type(16))) __bf16 v16bf;
typedef __attribute__((ext_vector_type(8)))  float  v8f;

#define DEV __device__ __forceinline__

DEV unsigned short f2bf(float f) {          // f32 -> bf16, round-to-nearest-even
  unsigned int u = __float_as_uint(f);
  u += 0x7FFFu + ((u >> 16) & 1u);
  return (unsigned short)(u >> 16);
}
DEV __bf16 usbf(unsigned short u) {
  union { unsigned short u; __bf16 b; } c; c.u = u; return c.b;
}
DEV v8f wmma_bf16(v16bf a, v16bf b, v8f c) {
  // v_wmma_f32_16x16x32_bf16: (neg_a, A, neg_b, B, c_mod, C, reuse_a, reuse_b)
  return __builtin_amdgcn_wmma_f32_16x16x32_bf16(false, a, false, b, (short)0, c,
                                                 false, false);
}

// ---- WMMA fragment loaders (CDNA5 16-bit layouts, wave32) -------------------
// 16x32 operand whose K dim is contiguous per lane (lda in u16 units):
// lane L: line = L&15, khalf = L>>4; VGPR v holds K pair
//   kb = (v&4 ? 16 : 0) + khalf*8 + 2*(v&3)
// Per lane this is two contiguous 8-element (16 B) runs -> 2x ds_load_b128.
// Used for A tiles (line = M row) and for B tiles stored transposed
// (line = N column, element (k,n) = s[n*lda + k]).
DEV v16bf frag_k_contig(const unsigned short* s, int lda, int lane) {
  int line = lane & 15, kh = lane >> 4;
  const unsigned short* p = s + line * lda;
  v16bf a;
#pragma unroll
  for (int v = 0; v < 8; ++v) {
    int kb = ((v & 4) ? 16 : 0) + kh * 8 + 2 * (v & 3);
    a[2 * v]     = usbf(p[kb]);
    a[2 * v + 1] = usbf(p[kb + 1]);
  }
  return a;
}

// ---- block reduction (wave32) ----------------------------------------------
DEV float blk_sum(float v, float* red8) {
#pragma unroll
  for (int o = 16; o > 0; o >>= 1) v += __shfl_down(v, o, 32);
  int tid = threadIdx.x;
  if ((tid & 31) == 0) red8[tid >> 5] = v;
  __syncthreads();
  if (tid == 0) {
    float s = 0.f;
#pragma unroll
    for (int i = 0; i < 8; ++i) s += red8[i];
    red8[0] = s;
  }
  __syncthreads();
  float r = red8[0];
  __syncthreads();
  return r;
}

// ============================================================================
// Embedding + LayerNorm: one block per token row (8192 rows), D = 768.
// ============================================================================
__global__ __launch_bounds__(256)
void embed_ln_kernel(const int* __restrict__ tok1, const int* __restrict__ tok2,
                     const float* __restrict__ word_emb,
                     const float* __restrict__ pos_emb,
                     const float* __restrict__ type_emb,
                     const float* __restrict__ g, const float* __restrict__ b,
                     float* __restrict__ H)
{
  __shared__ float red[8];
  int m    = blockIdx.x;        // 0..8191
  int enc  = m >> 12;           // which input tensor
  int rem  = m & 4095;          // b*2048 + s
  int srow = rem & 2047;
  int token = (enc ? tok2 : tok1)[rem];
  int tid = threadIdx.x;
  const float* we = word_emb + (size_t)token * 768;
  const float* pe = pos_emb + (size_t)srow * 768;
  float x[3];
  float s = 0.f;
#pragma unroll
  for (int i = 0; i < 3; ++i) {
    int d = tid + i * 256;
    x[i] = we[d] + pe[d] + type_emb[d];
    s += x[i];
  }
  s = blk_sum(s, red);
  float mean = s * (1.0f / 768.0f);
  float vs = 0.f;
#pragma unroll
  for (int i = 0; i < 3; ++i) { float d0 = x[i] - mean; vs += d0 * d0; }
  vs = blk_sum(vs, red);
  float inv = rsqrtf(vs * (1.0f / 768.0f) + 1e-5f);
#pragma unroll
  for (int i = 0; i < 3; ++i) {
    int d = tid + i * 256;
    H[(size_t)m * 768 + d] = (x[i] - mean) * inv * g[d] + b[d];
  }
}

// ============================================================================
// LayerNorm: Y = LN(X)*g + b, one block per row, D = 768.
// ============================================================================
__global__ __launch_bounds__(256)
void ln_kernel(const float* __restrict__ X, const float* __restrict__ g,
               const float* __restrict__ b, float* __restrict__ Y)
{
  __shared__ float red[8];
  int m = blockIdx.x, tid = threadIdx.x;
  const float* xr = X + (size_t)m * 768;
  float x[3];
  float s = 0.f;
#pragma unroll
  for (int i = 0; i < 3; ++i) { x[i] = xr[tid + i * 256]; s += x[i]; }
  s = blk_sum(s, red);
  float mean = s * (1.0f / 768.0f);
  float vs = 0.f;
#pragma unroll
  for (int i = 0; i < 3; ++i) { float d0 = x[i] - mean; vs += d0 * d0; }
  vs = blk_sum(vs, red);
  float inv = rsqrtf(vs * (1.0f / 768.0f) + 1e-5f);
#pragma unroll
  for (int i = 0; i < 3; ++i) {
    int d = tid + i * 256;
    Y[(size_t)m * 768 + d] = (x[i] - mean) * inv * g[d] + b[d];
  }
}

// ============================================================================
// Generic WMMA GEMM: C = act(A[MxK] @ W[KxN] + bias [+ residual])
// Block tile 128x128, 8 waves, each wave 32x64 (2x4 accumulators).
// Double-buffered LDS (one barrier per K step); B stored transposed (N-major)
// so B fragments load as ds_load_b128 pairs like A fragments.
// M%128==0, N%128==0, K%32==0. act: 0=none, 1=exact GELU.
// ============================================================================
__global__ __launch_bounds__(256)
void gemm_wmma_kernel(const float* __restrict__ A, const float* __restrict__ W,
                      const float* __restrict__ bias,
                      const float* __restrict__ residual,
                      float* __restrict__ C, int M, int N, int K, int act)
{
  __shared__ unsigned short As[2][128 * 32];   // row-major M x K tile (bf16)
  __shared__ unsigned short Bt[2][128 * 32];   // transposed: [n][k], stride 32
  int tid  = threadIdx.x;
  int lane = tid & 31;
  int wave = tid >> 5;                      // 0..7
  int wm = (wave & 3) * 32;                 // wave row offset in tile
  int wn = (wave >> 2) * 64;                // wave col offset in tile
  int m0 = blockIdx.y * 128;
  int n0 = blockIdx.x * 128;
  int nk = K >> 5;

  v8f acc[2][4];
#pragma unroll
  for (int i = 0; i < 2; ++i)
#pragma unroll
    for (int j = 0; j < 4; ++j) acc[i][j] = (v8f)0.0f;

  auto stage = [&](int buf, int k0) {
    // A tile: 128x32, float4 along K, 4 per thread
#pragma unroll
    for (int i = 0; i < 4; ++i) {
      int idx = tid + i * 256;
      int r = idx >> 3, c4 = (idx & 7) << 2;
      float4 f = *(const float4*)(A + (size_t)(m0 + r) * K + k0 + c4);
      unsigned short* d = &As[buf][r * 32 + c4];
      d[0] = f2bf(f.x); d[1] = f2bf(f.y); d[2] = f2bf(f.z); d[3] = f2bf(f.w);
    }
    // B tile transposed: thread owns column n, 4 consecutive k rows.
    // Global reads coalesce across lanes (consecutive n); LDS store is uint2.
#pragma unroll
    for (int i = 0; i < 4; ++i) {
      int idx = tid + i * 256;
      int n = idx & 127, rb = (idx >> 7) << 2;
      const float* wp = W + (size_t)(k0 + rb) * N + n0 + n;
      unsigned int u0 = (unsigned int)f2bf(wp[0]) |
                        ((unsigned int)f2bf(wp[(size_t)N]) << 16);
      unsigned int u1 = (unsigned int)f2bf(wp[2 * (size_t)N]) |
                        ((unsigned int)f2bf(wp[3 * (size_t)N]) << 16);
      uint2 uu; uu.x = u0; uu.y = u1;
      *(uint2*)(&Bt[buf][n * 32 + rb]) = uu;
    }
  };

  stage(0, 0);
  for (int kt = 0; kt < nk; ++kt) {
    __syncthreads();
    int cur = kt & 1;
    if (kt + 1 < nk) {
      __builtin_prefetch(A + (size_t)(m0 + (tid >> 1)) * K + ((kt + 1) << 5) +
                         ((tid & 1) << 4));
      stage(cur ^ 1, (kt + 1) << 5);
    }
    const unsigned short* Ab = As[cur];
    const unsigned short* Bb = Bt[cur];
    v16bf af0 = frag_k_contig(Ab + (wm + 0)  * 32, 32, lane);
    v16bf af1 = frag_k_contig(Ab + (wm + 16) * 32, 32, lane);
#pragma unroll
    for (int nt = 0; nt < 4; ++nt) {
      v16bf bf_ = frag_k_contig(Bb + (wn + nt * 16) * 32, 32, lane);
      acc[0][nt] = wmma_bf16(af0, bf_, acc[0][nt]);
      acc[1][nt] = wmma_bf16(af1, bf_, acc[1][nt]);
    }
  }

  // epilogue: bias (+residual) (+GELU)
  int col = lane & 15, kh = lane >> 4;
#pragma unroll
  for (int mt = 0; mt < 2; ++mt)
#pragma unroll
    for (int nt = 0; nt < 4; ++nt) {
      int gn = n0 + wn + nt * 16 + col;
      float bvv = bias ? bias[gn] : 0.f;
#pragma unroll
      for (int r = 0; r < 8; ++r) {
        int gm = m0 + wm + mt * 16 + kh * 8 + r;
        float v = acc[mt][nt][r] + bvv;
        if (residual) v += residual[(size_t)gm * N + gn];
        if (act == 1) v = 0.5f * v * (1.0f + erff(v * 0.70710678118654752f));
        C[(size_t)gm * N + gn] = v;
      }
    }
}

// ============================================================================
// Band attention: one workgroup per (seq, head, chunk); 4*12*16 = 768 blocks.
// 256 KB dynamic LDS (CDNA5 allows 320 KB/workgroup):
//   Qs  128x64 bf16 (16 KB)                       A operand, K-contiguous
//   KVs 48 KB: K band row-major 384x64 (Kt for scores is naturally
//              K-contiguous per lane), then V transposed 64x384
//   Ps  128x384 f32 (192 KB) score tile; converted in place to bf16 rows
//       after softmax so P fragments are K-contiguous too.
// ============================================================================
__global__ __launch_bounds__(256)
void band_attn_kernel(const float* __restrict__ Q, const float* __restrict__ Kv,
                      const float* __restrict__ Vv, float* __restrict__ O)
{
  extern __shared__ char smem[];
  unsigned short* Qs  = (unsigned short*)smem;                 // 128*64
  unsigned short* KVs = (unsigned short*)(smem + 16384);       // 384*64 / 64*384
  float*          Ps  = (float*)(smem + 65536);                // 128*384

  int bid   = blockIdx.x;
  int chunk = bid & 15;
  int head  = (bid >> 4) % 12;
  int seq   = (bid >> 4) / 12;
  int tid  = threadIdx.x;
  int lane = tid & 31;
  int wave = tid >> 5;
  size_t rowbase = (size_t)seq * 2048;

  // stage Q chunk (128x64, row-major)
#pragma unroll 4
  for (int i = 0; i < 8; ++i) {
    int idx = tid + i * 256;
    int r = idx >> 4, c4 = (idx & 15) << 2;
    float4 f = *(const float4*)(Q + (rowbase + (size_t)(chunk * 128 + r)) * 768 +
                                head * 64 + c4);
    Qs[r * 64 + c4 + 0] = f2bf(f.x);
    Qs[r * 64 + c4 + 1] = f2bf(f.y);
    Qs[r * 64 + c4 + 2] = f2bf(f.z);
    Qs[r * 64 + c4 + 3] = f2bf(f.w);
  }
  // stage K band (384x64 row-major == key-major -> B^T is K-contiguous)
#pragma unroll 4
  for (int i = 0; i < 24; ++i) {
    int idx = tid + i * 256;
    int r = idx >> 4, c4 = (idx & 15) << 2;
    int gk = chunk * 128 - 128 + r;
    float4 f = make_float4(0.f, 0.f, 0.f, 0.f);
    if (gk >= 0 && gk < 2048)
      f = *(const float4*)(Kv + (rowbase + (size_t)gk) * 768 + head * 64 + c4);
    KVs[r * 64 + c4 + 0] = f2bf(f.x);
    KVs[r * 64 + c4 + 1] = f2bf(f.y);
    KVs[r * 64 + c4 + 2] = f2bf(f.z);
    KVs[r * 64 + c4 + 3] = f2bf(f.w);
  }
  __syncthreads();

  // scores = Q @ K^T * scale, masked; each wave owns one 16-row M tile.
  {
    v16bf aq0 = frag_k_contig(Qs + (wave * 16) * 64, 64, lane);       // K 0..31
    v16bf aq1 = frag_k_contig(Qs + (wave * 16) * 64 + 32, 64, lane);  // K 32..63
    int col = lane & 15, kh = lane >> 4;
#pragma unroll
    for (int nt = 0; nt < 24; ++nt) {
      v16bf b0 = frag_k_contig(KVs + (nt * 16) * 64, 64, lane);
      v16bf b1 = frag_k_contig(KVs + (nt * 16) * 64 + 32, 64, lane);
      v8f a = (v8f)0.0f;
      a = wmma_bf16(aq0, b0, a);
      a = wmma_bf16(aq1, b1, a);
      int kj = nt * 16 + col;
      int gk = chunk * 128 - 128 + kj;
#pragma unroll
      for (int r = 0; r < 8; ++r) {
        int qi = wave * 16 + kh * 8 + r;
        bool ok = (kj >= qi) && (kj <= qi + 256) && (gk >= 0) && (gk < 2048);
        Ps[qi * 384 + kj] = ok ? a[r] * 0.125f : -1.0e9f;
      }
    }
  }
  __syncthreads();

  // row softmax over 384 keys (threads 0..127, one row each), then convert the
  // row to bf16 IN PLACE over the first half of its f32 storage (safe forward
  // overwrite: ushort j lands in float j/2 <= already-consumed floats).
  if (tid < 128) {
    float* row = Ps + tid * 384;
    unsigned short* brow = (unsigned short*)row;
    float mx = -3.0e38f;
    for (int j = 0; j < 384; ++j) mx = fmaxf(mx, row[j]);
    float s = 0.f;
    for (int j = 0; j < 384; ++j) { float e = __expf(row[j] - mx); row[j] = e; s += e; }
    float inv = 1.0f / s;
    for (int j = 0; j < 384; ++j) {
      float p = row[j] * inv;
      brow[j] = f2bf(p);
    }
  }
  __syncthreads();

  // stage V band TRANSPOSED over the K buffer: Vt[d][key], stride 384.
  // Thread owns dh column d and 4 consecutive key rows; uint2 LDS store.
#pragma unroll 4
  for (int i = 0; i < 24; ++i) {
    int idx = tid + i * 256;          // 0..6143
    int d  = idx & 63;
    int rb = (idx >> 6) << 2;         // key row base 0..380
    int gk = chunk * 128 - 128 + rb;
    float v0 = 0.f, v1 = 0.f, v2 = 0.f, v3 = 0.f;
    const float* vp = Vv + (rowbase + (size_t)gk) * 768 + head * 64 + d;
    if (gk + 0 >= 0 && gk + 0 < 2048) v0 = vp[0];
    if (gk + 1 >= 0 && gk + 1 < 2048) v1 = vp[768];
    if (gk + 2 >= 0 && gk + 2 < 2048) v2 = vp[2 * 768];
    if (gk + 3 >= 0 && gk + 3 < 2048) v3 = vp[3 * 768];
    unsigned int u0 = (unsigned int)f2bf(v0) | ((unsigned int)f2bf(v1) << 16);
    unsigned int u1 = (unsigned int)f2bf(v2) | ((unsigned int)f2bf(v3) << 16);
    uint2 uu; uu.x = u0; uu.y = u1;
    *(uint2*)(&KVs[d * 384 + rb]) = uu;
  }
  __syncthreads();

  // out = P(128x384) @ V(384x64); wave owns its 16-row tile, 4 N tiles, 12 K steps
  {
    v8f accO[4];
#pragma unroll
    for (int nt = 0; nt < 4; ++nt) accO[nt] = (v8f)0.0f;
    const unsigned short* Pb = (const unsigned short*)Ps;   // bf16 rows, lda=768
    for (int kt = 0; kt < 12; ++kt) {
      v16bf ap = frag_k_contig(Pb + (wave * 16) * 768 + kt * 32, 768, lane);
#pragma unroll
      for (int nt = 0; nt < 4; ++nt) {
        v16bf bv = frag_k_contig(KVs + (nt * 16) * 384 + kt * 32, 384, lane);
        accO[nt] = wmma_bf16(ap, bv, accO[nt]);
      }
    }
    int col = lane & 15, kh = lane >> 4;
#pragma unroll
    for (int nt = 0; nt < 4; ++nt)
#pragma unroll
      for (int r = 0; r < 8; ++r) {
        int qi = wave * 16 + kh * 8 + r;
        O[(rowbase + (size_t)(chunk * 128 + qi)) * 768 + head * 64 + nt * 16 + col] =
            accO[nt][r];
      }
  }
}

// ============================================================================
// Pooler: pooled[s] = tanh(h[s,0,:] @ pool_w + pool_b); 4 blocks.
// ============================================================================
__global__ __launch_bounds__(256)
void pool_kernel(const float* __restrict__ H, const float* __restrict__ pw,
                 const float* __restrict__ pb, float* __restrict__ pooled)
{
  __shared__ float hrow[768];
  int s = blockIdx.x, tid = threadIdx.x;
  const float* hr = H + (size_t)s * 2048 * 768;
  for (int i = tid; i < 768; i += 256) hrow[i] = hr[i];
  __syncthreads();
  for (int c = tid; c < 768; c += 256) {
    float acc = pb[c];
    for (int k = 0; k < 768; ++k) acc += hrow[k] * pw[(size_t)k * 768 + c];
    pooled[s * 768 + c] = tanhf(acc);
  }
}

// ============================================================================
// Head: x[b] = [pooled[b], pooled[2+b]] (1536); y = relu(x@head_w + head_b);
// out = y @ out_w -> (2,2). Single block.
// ============================================================================
__global__ __launch_bounds__(256)
void head_kernel(const float* __restrict__ pooled, const float* __restrict__ hw,
                 const float* __restrict__ hb, const float* __restrict__ ow,
                 float* __restrict__ out)
{
  __shared__ float ys[2][256];
  __shared__ float red[8];
  int j = threadIdx.x;
  for (int b = 0; b < 2; ++b) {
    float acc = hb[j];
    for (int k = 0; k < 768; ++k) acc += pooled[b * 768 + k] * hw[(size_t)k * 256 + j];
    for (int k = 0; k < 768; ++k)
      acc += pooled[(2 + b) * 768 + k] * hw[(size_t)(768 + k) * 256 + j];
    ys[b][j] = fmaxf(acc, 0.f);
  }
  __syncthreads();
  for (int b = 0; b < 2; ++b)
    for (int o = 0; o < 2; ++o) {
      float p = ys[b][j] * ow[j * 2 + o];
      p = blk_sum(p, red);
      if (j == 0) out[b * 2 + o] = p;
      __syncthreads();
    }
}

// ============================================================================
// Host-side orchestration
// ============================================================================
extern "C" void kernel_launch(void* const* d_in, const int* in_sizes, int n_in,
                              void* d_out, int out_size, void* d_ws, size_t ws_size,
                              hipStream_t stream)
{
  if (n_in < 28) return;
  const int*   tok1     = (const int*)d_in[0];
  const int*   tok2     = (const int*)d_in[1];
  const float* emb_ln_b = (const float*)d_in[2];
  const float* emb_ln_s = (const float*)d_in[3];
  const float* head_b   = (const float*)d_in[4];
  const float* head_w   = (const float*)d_in[5];
  const float* b1       = (const float*)d_in[6];
  const float* b2       = (const float*)d_in[7];
  const float* bk       = (const float*)d_in[8];
  const float* bo       = (const float*)d_in[9];
  const float* bq       = (const float*)d_in[10];
  const float* bv       = (const float*)d_in[11];
  const float* ln1_b    = (const float*)d_in[12];
  const float* ln1_s    = (const float*)d_in[13];
  const float* ln2_b    = (const float*)d_in[14];
  const float* ln2_s    = (const float*)d_in[15];
  const float* w1       = (const float*)d_in[16];
  const float* w2       = (const float*)d_in[17];
  const float* wk       = (const float*)d_in[18];
  const float* wo       = (const float*)d_in[19];
  const float* wq       = (const float*)d_in[20];
  const float* wv       = (const float*)d_in[21];
  const float* out_w    = (const float*)d_in[22];
  const float* pool_b   = (const float*)d_in[23];
  const float* pool_w   = (const float*)d_in[24];
  const float* pos_emb  = (const float*)d_in[25];
  const float* type_emb = (const float*)d_in[26];
  const float* word_emb = (const float*)d_in[27];

  float* ws = (float*)d_ws;
  const size_t BUF = (size_t)8192 * 768;     // 2048*3072 == 8192*768
  float* h      = ws;
  float* qb     = ws + 1 * BUF;
  float* kb     = ws + 2 * BUF;
  float* vb     = ws + 3 * BUF;
  float* ab     = ws + 4 * BUF;
  float* tmp    = ws + 5 * BUF;
  float* ff     = ws + 6 * BUF;              // 2048 x 3072 FFN chunk scratch
  float* pooled = ws + 7 * BUF;              // 4 x 768

  // band_attn needs 256 KB dynamic LDS (not a stream op; capture-safe)
  (void)hipFuncSetAttribute((const void*)band_attn_kernel,
                            hipFuncAttributeMaxDynamicSharedMemorySize, 262144);

  embed_ln_kernel<<<8192, 256, 0, stream>>>(tok1, tok2, word_emb, pos_emb,
                                            type_emb, emb_ln_s, emb_ln_b, h);

  for (int l = 0; l < 4; ++l) {
    const float* wq_l = wq + (size_t)l * 768 * 768;
    const float* wk_l = wk + (size_t)l * 768 * 768;
    const float* wv_l = wv + (size_t)l * 768 * 768;
    const float* wo_l = wo + (size_t)l * 768 * 768;
    const float* w1_l = w1 + (size_t)l * 768 * 3072;
    const float* w2_l = w2 + (size_t)l * 3072 * 768;
    const float* bq_l = bq + l * 768;
    const float* bk_l = bk + l * 768;
    const float* bv_l = bv + l * 768;
    const float* bo_l = bo + l * 768;
    const float* b1_l = b1 + l * 3072;
    const float* b2_l = b2 + l * 768;

    gemm_wmma_kernel<<<dim3(6, 64), 256, 0, stream>>>(h, wq_l, bq_l, nullptr, qb,
                                                      8192, 768, 768, 0);
    gemm_wmma_kernel<<<dim3(6, 64), 256, 0, stream>>>(h, wk_l, bk_l, nullptr, kb,
                                                      8192, 768, 768, 0);
    gemm_wmma_kernel<<<dim3(6, 64), 256, 0, stream>>>(h, wv_l, bv_l, nullptr, vb,
                                                      8192, 768, 768, 0);
    band_attn_kernel<<<768, 256, 262144, stream>>>(qb, kb, vb, ab);
    gemm_wmma_kernel<<<dim3(6, 64), 256, 0, stream>>>(ab, wo_l, bo_l, h, tmp,
                                                      8192, 768, 768, 0);
    ln_kernel<<<8192, 256, 0, stream>>>(tmp, ln1_s + l * 768, ln1_b + l * 768, h);

    for (int mc = 0; mc < 4; ++mc) {                 // FFN in 2048-row chunks
      const float* hc = h + (size_t)mc * 2048 * 768;
      gemm_wmma_kernel<<<dim3(24, 16), 256, 0, stream>>>(hc, w1_l, b1_l, nullptr,
                                                         ff, 2048, 3072, 768, 1);
      gemm_wmma_kernel<<<dim3(6, 16), 256, 0, stream>>>(ff, w2_l, b2_l, hc,
                                                        tmp + (size_t)mc * 2048 * 768,
                                                        2048, 768, 3072, 0);
    }
    ln_kernel<<<8192, 256, 0, stream>>>(tmp, ln2_s + l * 768, ln2_b + l * 768, h);
  }

  pool_kernel<<<4, 256, 0, stream>>>(h, pool_w, pool_b, pooled);
  head_kernel<<<1, 256, 0, stream>>>(pooled, head_w, head_b, out_w, (float*)d_out);
}
